// WindowAttentionFlex_3788161155763
// MI455X (gfx1250) — matrix operand order repaired and checked by reading, not compiled, not run on hardware
//
#include <hip/hip_runtime.h>
#include <math.h>

// ---------------------------------------------------------------------------
// WindowAttentionFlex for MI455X (gfx1250, wave32, WMMA)
// Pipeline: f32->bf16 converts | QKV GEMM (wmma bf16) | fused norm+RoPE+
// flash-style window attention (wmma bf16) | proj GEMM (wmma bf16, f32 out)
// ---------------------------------------------------------------------------

#define DIMC   512
#define HEADS  16
#define HD     32
#define NWIN   64
#define NTOK   64           // tokens per window (8x8)
#define BATCH  8
#define QKVC   1536
#define LN100  4.605170185988092f
#define RSQRT_HD 0.17677669529663687f  // 1/sqrt(32)

typedef __attribute__((ext_vector_type(16))) __bf16 v16bf;
typedef __attribute__((ext_vector_type(8)))  float  v8f;

union ABFrag { v16bf v; uint4 q[2]; unsigned short s[16]; };

__device__ __forceinline__ unsigned short f2bf(float f) {
    union { float f; unsigned u; } v; v.f = f;
    unsigned r = v.u + 0x7FFFu + ((v.u >> 16) & 1u);   // round-to-nearest-even
    return (unsigned short)(r >> 16);
}
__device__ __forceinline__ float bf2f(unsigned short h) {
    union { unsigned u; float f; } v; v.u = ((unsigned)h) << 16;
    return v.f;
}

// A-matrix fragment (16x32 bf16, row-major source, leading dim ld).
// Lane L: M=L&15, halves i -> K = kb + i,
// kb = (L>=16)*16.  (contiguous K order, matching the B loaders)
__device__ __forceinline__ v16bf load_a16(const unsigned short* src, int ld,
                                          int row0, int k0, int lane) {
    int row = row0 + (lane & 15);
    int kb  = k0 + ((lane >> 4) << 4);
    const unsigned short* p = src + (size_t)row * ld + kb;
    ABFrag f;
    f.q[0] = *(const uint4*)(p);
    f.q[1] = *(const uint4*)(p + 8);
    return f.v;
}

// B-matrix fragment (32x16 bf16) where B[k][n] = src[n][k] (src row-major NxK,
// i.e. B = src^T: weight rows / k rows as columns).
// Lane L: N=L&15, halves i -> K = (L>=16)*16 + i.  (contiguous K order, same as load_a16)
__device__ __forceinline__ v16bf load_bT16(const unsigned short* src, int ld,
                                           int col0, int k0, int lane) {
    int col = col0 + (lane & 15);
    int kb  = k0 + ((lane >> 4) << 4);
    const unsigned short* p = src + (size_t)col * ld + kb;
    ABFrag f;
    f.q[0] = *(const uint4*)(p);
    f.q[1] = *(const uint4*)(p + 8);
    return f.v;
}

// B-matrix fragment where B is stored row-major KxN (e.g. v: 64x32 in LDS).
// b[i] = src[k0 + (lane>=16)*16 + i][col0 + (lane&15)]  (strided gather)
__device__ __forceinline__ v16bf load_bKN16(const unsigned short* src, int ld,
                                            int col0, int k0, int lane) {
    int col = col0 + (lane & 15);
    int kb  = k0 + ((lane >> 4) << 4);
    ABFrag f;
#pragma unroll
    for (int i = 0; i < 16; ++i)
        f.s[i] = src[(size_t)(kb + i) * ld + col];
    return f.v;
}

// ---------------------------------------------------------------------------
// f32 -> bf16 bulk convert (4 elements/thread)
// ---------------------------------------------------------------------------
__global__ void f32_to_bf16_kernel(const float* __restrict__ in,
                                   unsigned short* __restrict__ out, int n4) {
    int i = blockIdx.x * blockDim.x + threadIdx.x;
    if (i < n4) {
        float4 v = ((const float4*)in)[i];
        ushort4 o;
        o.x = f2bf(v.x); o.y = f2bf(v.y); o.z = f2bf(v.z); o.w = f2bf(v.w);
        ((ushort4*)out)[i] = o;
    }
}

// ---------------------------------------------------------------------------
// QKV GEMM: qkv[m][n] = x[m][:] . W[n][:] + bias(n),  M=32768 N=1536 K=512
// block = 256 thr (8 waves), block tile 64x128, wave tile 32x32 (2x2 wmma)
// ---------------------------------------------------------------------------
__global__ __launch_bounds__(256) void qkv_gemm_kernel(
    const unsigned short* __restrict__ xbf,
    const unsigned short* __restrict__ wbf,
    const float* __restrict__ qbias, const float* __restrict__ vbias,
    unsigned short* __restrict__ out) {
    const int lane = threadIdx.x & 31;
    const int wv   = threadIdx.x >> 5;
    const int m0 = blockIdx.y * 64  + (wv & 1) * 32;
    const int n0 = blockIdx.x * 128 + (wv >> 1) * 32;

    v8f acc00 = {}, acc01 = {}, acc10 = {}, acc11 = {};
    for (int k0 = 0; k0 < DIMC; k0 += 32) {
        // prefetch next K-tile of x into caches (global_prefetch_b8)
        __builtin_prefetch(xbf + (size_t)(m0 + (lane & 15)) * DIMC + k0 + 64, 0, 0);
        v16bf a0 = load_a16(xbf, DIMC, m0,      k0, lane);
        v16bf a1 = load_a16(xbf, DIMC, m0 + 16, k0, lane);
        v16bf b0 = load_bT16(wbf, DIMC, n0,      k0, lane);
        v16bf b1 = load_bT16(wbf, DIMC, n0 + 16, k0, lane);
        acc00 = __builtin_amdgcn_wmma_f32_16x16x32_bf16(false, a0, false, b0, (short)0, acc00, false, false);
        acc01 = __builtin_amdgcn_wmma_f32_16x16x32_bf16(false, a0, false, b1, (short)0, acc01, false, false);
        acc10 = __builtin_amdgcn_wmma_f32_16x16x32_bf16(false, a1, false, b0, (short)0, acc10, false, false);
        acc11 = __builtin_amdgcn_wmma_f32_16x16x32_bf16(false, a1, false, b1, (short)0, acc11, false, false);
    }

    const int col = lane & 15, rg = (lane >> 4) * 8;
#pragma unroll
    for (int tm = 0; tm < 2; ++tm) {
#pragma unroll
        for (int tn = 0; tn < 2; ++tn) {
            const v8f acc = (tm == 0) ? (tn == 0 ? acc00 : acc01)
                                      : (tn == 0 ? acc10 : acc11);
            int ccol = n0 + tn * 16 + col;
            float bias = (ccol < DIMC) ? qbias[ccol]
                       : (ccol < 2 * DIMC) ? 0.0f : vbias[ccol - 2 * DIMC];
#pragma unroll
            for (int r = 0; r < 8; ++r) {
                int row = m0 + tm * 16 + rg + r;
                out[(size_t)row * QKVC + ccol] = f2bf(acc[r] + bias);
            }
        }
    }
}

// ---------------------------------------------------------------------------
// Fused window attention: one block per (b, h, window). 128 thr = 4 waves.
// Loads q/k (64x32 bf16), L2-normalize, logit-scale (q, with 1/sqrt(HD)
// folded in), axial RoPE; scores via wmma (K=32=HD in one op), shuffle
// softmax, attn@v via wmma with f32 accumulation across K=64.
// ---------------------------------------------------------------------------
__global__ __launch_bounds__(128) void win_attn_kernel(
    const unsigned short* __restrict__ qkv,
    const float* __restrict__ logit_scale,
    unsigned short* __restrict__ attn_out) {
    const int idx = blockIdx.x;
    const int w = idx & 63, h = (idx >> 6) & 15, b = idx >> 10;

    __shared__ unsigned short qs[NTOK][HD];
    __shared__ unsigned short ks[NTOK][HD];
    __shared__ unsigned short vs[NTOK][HD];
    __shared__ unsigned short attn_s[4][16][NTOK];

    const int t = threadIdx.x, lane = t & 31, wv = t >> 5;
    const size_t tokbase = ((size_t)(b * NWIN + w)) * NTOK;

    // ---- load + normalize + scale + RoPE (threads 0-63: q, 64-127: k) ----
    {
        const int  tok = t & 63;
        const bool isq = t < 64;
        const size_t rowoff = (tokbase + tok) * QKVC + (isq ? 0 : DIMC) + h * HD;
        const uint4* p = (const uint4*)(qkv + rowoff);   // 32 bf16 = 4x uint4
        ABFrag u0, u1;
        u0.q[0] = p[0]; u0.q[1] = p[1]; u1.q[0] = p[2]; u1.q[1] = p[3];
        float x[HD];
#pragma unroll
        for (int i = 0; i < 16; ++i) { x[i] = bf2f(u0.s[i]); x[16 + i] = bf2f(u1.s[i]); }
        float ss = 0.f;
#pragma unroll
        for (int i = 0; i < HD; ++i) ss += x[i] * x[i];
        float rn = 1.0f / fmaxf(sqrtf(ss), 1e-12f);
        float sc = isq ? __expf(fmaxf(logit_scale[h], LN100)) * rn * RSQRT_HD : rn;
        const int ih = tok >> 3, iw = tok & 7;
        unsigned short* dst = isq ? &qs[tok][0] : &ks[tok][0];
#pragma unroll
        for (int pr = 0; pr < 16; ++pr) {     // interleaved rotary pairs
            float pos = (float)((pr < 8) ? ih : iw);
            float f = pos * __powf(10000.0f, -(float)(pr & 7) * 0.125f);
            float s, c; __sincosf(f, &s, &c);
            float x0 = x[2 * pr] * sc, x1 = x[2 * pr + 1] * sc;
            dst[2 * pr]     = f2bf(x0 * c - x1 * s);
            dst[2 * pr + 1] = f2bf(x1 * c + x0 * s);
        }
        // v: raw bf16 copy, 16 elems (32B) per thread
        const int vtok = t >> 1, half = t & 1;
        const uint4* pv = (const uint4*)(qkv + (tokbase + vtok) * QKVC + 2 * DIMC + h * HD + half * 16);
        *(uint4*)&vs[vtok][half * 16]     = pv[0];
        *(uint4*)&vs[vtok][half * 16 + 8] = pv[1];
    }
    __syncthreads();

    // ---- scores: wave wv owns query rows [wv*16, wv*16+16) x 64 cols ----
    v8f z = {};
    v8f scor[4];
    {
        v16bf aq = load_a16(&qs[0][0], HD, wv * 16, 0, lane);
#pragma unroll
        for (int tn = 0; tn < 4; ++tn) {
            v16bf bk = load_bT16(&ks[0][0], HD, tn * 16, 0, lane);  // B = k^T
            scor[tn] = __builtin_amdgcn_wmma_f32_16x16x32_bf16(false, aq, false, bk, (short)0, z, false, false);
        }
    }

    // ---- softmax across 64 keys (4 tiles x 16 lanes of one half-wave) ----
    const int colq = lane & 15, rg = (lane >> 4) * 8;
#pragma unroll
    for (int r = 0; r < 8; ++r) {
        float m = fmaxf(fmaxf(scor[0][r], scor[1][r]), fmaxf(scor[2][r], scor[3][r]));
        for (int off = 8; off >= 1; off >>= 1) m = fmaxf(m, __shfl_xor(m, off, 32));
        float s = 0.f;
#pragma unroll
        for (int tn = 0; tn < 4; ++tn) { scor[tn][r] = __expf(scor[tn][r] - m); s += scor[tn][r]; }
        for (int off = 8; off >= 1; off >>= 1) s += __shfl_xor(s, off, 32);
        float inv = 1.0f / s;
#pragma unroll
        for (int tn = 0; tn < 4; ++tn)
            attn_s[wv][rg + r][tn * 16 + colq] = f2bf(scor[tn][r] * inv);
    }
    // per-wave LDS region; DS ops are in-order within a wave -> no barrier

    // ---- out = attn(16x64) @ v(64x32), K in two 32-chunks, f32 accum ----
    v8f o0 = {}, o1 = {};
#pragma unroll
    for (int k0 = 0; k0 < NTOK; k0 += 32) {
        v16bf aa = load_a16(&attn_s[wv][0][0], NTOK, 0, k0, lane);
        v16bf bv0 = load_bKN16(&vs[0][0], HD, 0,  k0, lane);   // B row-major KxN
        v16bf bv1 = load_bKN16(&vs[0][0], HD, 16, k0, lane);
        o0 = __builtin_amdgcn_wmma_f32_16x16x32_bf16(false, aa, false, bv0, (short)0, o0, false, false);
        o1 = __builtin_amdgcn_wmma_f32_16x16x32_bf16(false, aa, false, bv1, (short)0, o1, false, false);
    }

    // ---- store (B, Nw, N, H*HD) bf16, channel = h*32 + d ----
#pragma unroll
    for (int r = 0; r < 8; ++r) {
        int qrow = wv * 16 + rg + r;
        size_t base = (tokbase + qrow) * DIMC + h * HD;
        attn_out[base + colq]      = f2bf(o0[r]);
        attn_out[base + 16 + colq] = f2bf(o1[r]);
    }
}

// ---------------------------------------------------------------------------
// Proj GEMM: out[m][n] = a[m][:] . W[n][:] + bias(n), M=32768 N=512 K=512
// fp32 output straight to d_out.
// ---------------------------------------------------------------------------
__global__ __launch_bounds__(256) void proj_gemm_kernel(
    const unsigned short* __restrict__ abf,
    const unsigned short* __restrict__ wbf,
    const float* __restrict__ bias,
    float* __restrict__ out) {
    const int lane = threadIdx.x & 31;
    const int wv   = threadIdx.x >> 5;
    const int m0 = blockIdx.y * 64  + (wv & 1) * 32;
    const int n0 = blockIdx.x * 128 + (wv >> 1) * 32;

    v8f acc00 = {}, acc01 = {}, acc10 = {}, acc11 = {};
    for (int k0 = 0; k0 < DIMC; k0 += 32) {
        __builtin_prefetch(abf + (size_t)(m0 + (lane & 15)) * DIMC + k0 + 64, 0, 0);
        v16bf a0 = load_a16(abf, DIMC, m0,      k0, lane);
        v16bf a1 = load_a16(abf, DIMC, m0 + 16, k0, lane);
        v16bf b0 = load_bT16(wbf, DIMC, n0,      k0, lane);
        v16bf b1 = load_bT16(wbf, DIMC, n0 + 16, k0, lane);
        acc00 = __builtin_amdgcn_wmma_f32_16x16x32_bf16(false, a0, false, b0, (short)0, acc00, false, false);
        acc01 = __builtin_amdgcn_wmma_f32_16x16x32_bf16(false, a0, false, b1, (short)0, acc01, false, false);
        acc10 = __builtin_amdgcn_wmma_f32_16x16x32_bf16(false, a1, false, b0, (short)0, acc10, false, false);
        acc11 = __builtin_amdgcn_wmma_f32_16x16x32_bf16(false, a1, false, b1, (short)0, acc11, false, false);
    }

    const int col = lane & 15, rg = (lane >> 4) * 8;
#pragma unroll
    for (int tm = 0; tm < 2; ++tm) {
#pragma unroll
        for (int tn = 0; tn < 2; ++tn) {
            const v8f acc = (tm == 0) ? (tn == 0 ? acc00 : acc01)
                                      : (tn == 0 ? acc10 : acc11);
            int ccol = n0 + tn * 16 + col;
            float bv = bias[ccol];
#pragma unroll
            for (int r = 0; r < 8; ++r) {
                int row = m0 + tm * 16 + rg + r;
                out[(size_t)row * DIMC + ccol] = acc[r] + bv;
            }
        }
    }
}

// ---------------------------------------------------------------------------
// Host launcher
// ---------------------------------------------------------------------------
extern "C" void kernel_launch(void* const* d_in, const int* in_sizes, int n_in,
                              void* d_out, int out_size, void* d_ws, size_t ws_size,
                              hipStream_t stream) {
    const float* x       = (const float*)d_in[0];   // (8,64,64,512)
    const float* qkv_w   = (const float*)d_in[1];   // (1536,512)
    const float* q_bias  = (const float*)d_in[2];   // (512,)
    const float* v_bias  = (const float*)d_in[3];   // (512,)
    const float* lscale  = (const float*)d_in[4];   // (16,)
    const float* proj_w  = (const float*)d_in[5];   // (512,512)
    const float* proj_b  = (const float*)d_in[6];   // (512,)
    float* out = (float*)d_out;

    const size_t M = (size_t)BATCH * NWIN * NTOK;   // 32768 tokens

    // workspace layout (bf16 = unsigned short); attn_out reuses x_bf region
    unsigned short* x_bf    = (unsigned short*)d_ws;             // M*512
    unsigned short* qkvw_bf = x_bf    + M * DIMC;                // 1536*512
    unsigned short* projw_bf= qkvw_bf + (size_t)QKVC * DIMC;     // 512*512
    unsigned short* qkv_bf  = projw_bf+ (size_t)DIMC * DIMC;     // M*1536
    unsigned short* atto_bf = x_bf;                              // reuse (M*512)

    // 1) converts
    {
        int n4 = (int)(M * DIMC / 4);
        f32_to_bf16_kernel<<<(n4 + 255) / 256, 256, 0, stream>>>(x, x_bf, n4);
        n4 = QKVC * DIMC / 4;
        f32_to_bf16_kernel<<<(n4 + 255) / 256, 256, 0, stream>>>(qkv_w, qkvw_bf, n4);
        n4 = DIMC * DIMC / 4;
        f32_to_bf16_kernel<<<(n4 + 255) / 256, 256, 0, stream>>>(proj_w, projw_bf, n4);
    }
    // 2) QKV GEMM (+bias)
    qkv_gemm_kernel<<<dim3(QKVC / 128, (unsigned)(M / 64)), 256, 0, stream>>>(
        x_bf, qkvw_bf, q_bias, v_bias, qkv_bf);
    // 3) fused norm + RoPE + window attention
    win_attn_kernel<<<BATCH * HEADS * NWIN, 128, 0, stream>>>(qkv_bf, lscale, atto_bf);
    // 4) output projection (+bias) -> fp32
    proj_gemm_kernel<<<dim3(DIMC / 128, (unsigned)(M / 64)), 256, 0, stream>>>(
        atto_bf, projw_bf, proj_b, out);
}
